// MatrixScan_42606075576710
// MI455X (gfx1250) — matrix operand order repaired and checked by reading, not compiled, not run on hardware
//
#include <hip/hip_runtime.h>
#include <hip/hip_bf16.h>

#define B_  8
#define T_  4096
#define D_  1024
#define DS_ 64
#define C_  64
#define NC_ (T_ / C_)

typedef __attribute__((ext_vector_type(16))) _Float16 v16h;
typedef __attribute__((ext_vector_type(8)))  float    v8f;
typedef __attribute__((ext_vector_type(2)))  float    v2f;

// ---------------- WMMA wrappers ----------------
__device__ __forceinline__ v8f wmma_f16(v16h a, v16h b, v8f c) {
  return __builtin_amdgcn_wmma_f32_16x16x32_f16(false, a, false, b, (short)0, c,
                                                false, false);
}
__device__ __forceinline__ v8f wmma_f32(v2f a, v2f b, v8f c) {
  return __builtin_amdgcn_wmma_f32_16x16x4_f32(false, a, false, b, (short)0, c,
                                               false, false);
}

// ---------------- fragment loaders (wave32; f16 loaders use contiguous k = 16h+e in both A and B) ----------------
// A 16x32 f16 (MxK): lanes0-15 row M=lane, K 0..15; lanes16-31 K 16..31
__device__ __forceinline__ v16h afrag16(const _Float16* src, int ld, int k0) {
  const int lane = threadIdx.x & 31;
  const int m = lane & 15, h = lane >> 4;
  v16h a;
#pragma unroll
  for (int e = 0; e < 16; ++e) {
    const int k = k0 + h * 16 + e;
    a[e] = src[m * ld + k];
  }
  return a;
}
// B 32x16 f16 (KxN): lanes0-15 col N=lane K=0..15; lanes16-31 K=16..31
__device__ __forceinline__ v16h bfrag16(const _Float16* src, int ld, int k0, int n0) {
  const int lane = threadIdx.x & 31;
  const int n = lane & 15, h = lane >> 4;
  v16h b;
#pragma unroll
  for (int e = 0; e < 16; ++e) {
    const int k = k0 + h * 16 + e;
    b[e] = src[k * ld + n0 + n];
  }
  return b;
}
// A 16x4 f32: lanes0-15 row M=lane K=0,1; lanes16-31 K=2,3
__device__ __forceinline__ v2f afrag32(const float* src, int ld, int k0) {
  const int lane = threadIdx.x & 31;
  const int m = lane & 15, h = lane >> 4;
  v2f a;
  a.x = src[m * ld + k0 + h * 2];
  a.y = src[m * ld + k0 + h * 2 + 1];
  return a;
}
// B 4x16 f32: lanes0-15 col N=lane K=0,1; lanes16-31 K=2,3
__device__ __forceinline__ v2f bfrag32(const float* src, int ld, int k0, int n0) {
  const int lane = threadIdx.x & 31;
  const int n = lane & 15, h = lane >> 4;
  v2f b;
  b.x = src[(k0 + h * 2) * ld + n0 + n];
  b.y = src[(k0 + h * 2 + 1) * ld + n0 + n];
  return b;
}

__device__ __forceinline__ float sigmoidf_(float x) {
  return 1.0f / (1.0f + __expf(-x));
}

// ================= Kernel 1: fused q/k/v projection (WMMA f16) =================
// grid = BT/16 blocks of 128 threads (4 waves). Wave w owns n-tile w (0..3) for
// ALL THREE projections -> every weight/bias/output pointer is compile-time,
// so the compiler emits global (not flat) memory ops.
__global__ __launch_bounds__(128) void qkv_kernel(
    const float* __restrict__ x, const float* __restrict__ Wq,
    const float* __restrict__ bq, const float* __restrict__ Wk,
    const float* __restrict__ bk, const float* __restrict__ Wv,
    const float* __restrict__ bv, float* __restrict__ q, float* __restrict__ k,
    float* __restrict__ v) {
  __shared__ _Float16 xs[16][D_ + 8];
  const int r0 = blockIdx.x * 16;
  // vectorized LDS fill: 16x1024 f32 tile, float4 per step
  for (int idx = threadIdx.x; idx < (16 * D_) / 4; idx += 128) {
    const int m = idx >> 8;            // (idx*4) / 1024
    const int kk = (idx & 255) * 4;
    const float4 xv = *(const float4*)&x[(size_t)(r0 + m) * D_ + kk];
    xs[m][kk + 0] = (_Float16)xv.x;
    xs[m][kk + 1] = (_Float16)xv.y;
    xs[m][kk + 2] = (_Float16)xv.z;
    xs[m][kk + 3] = (_Float16)xv.w;
  }
  __syncthreads();

  const int nt = threadIdx.x >> 5;  // wave id == n-tile
  const int lane = threadIdx.x & 31;
  const int nn = lane & 15, h = lane >> 4;
  const size_t wrow = (size_t)(nt * 16 + nn) * D_;  // B[kk][n] = W[n_glob][kk]

  v8f aq = {}, ak = {}, av = {};
  for (int k0 = 0; k0 < D_; k0 += 32) {
    const v16h a = afrag16(&xs[0][0], D_ + 8, k0);
    v16h bqf, bkf, bvf;
#pragma unroll
    for (int e = 0; e < 16; ++e) {
      const int kk = k0 + h * 16 + e;  // contiguous over e -> b128 loads
      bqf[e] = (_Float16)Wq[wrow + kk];
      bkf[e] = (_Float16)Wk[wrow + kk];
      bvf[e] = (_Float16)Wv[wrow + kk];
    }
    aq = wmma_f16(a, bqf, aq);
    ak = wmma_f16(a, bkf, ak);
    av = wmma_f16(a, bvf, av);
  }
  const float biasq = bq[nt * 16 + nn];
  const float biask = bk[nt * 16 + nn];
  const float biasv = bv[nt * 16 + nn];
#pragma unroll
  for (int g = 0; g < 8; ++g) {
    const int m = g + (h << 3);
    const size_t o = (size_t)(r0 + m) * DS_ + nt * 16 + nn;
    q[o] = aq[g] + biasq;
    k[o] = ak[g] + biask;
    v[o] = av[g] + biasv;
  }
}

// ================= Kernel 2: per-chunk state contribution G =================
// G_c[i,j] = sum_u d_i^{C-1-u} v[u,i] k[u,j]  (all scales <= 1 -> f16 safe)
__global__ __launch_bounds__(128) void chunkG_kernel(
    const float* __restrict__ k, const float* __restrict__ v,
    const float* __restrict__ decay, float* __restrict__ G) {
  __shared__ _Float16 aT[DS_][C_ + 8];  // A[i][u] = d_i^{63-u} v[u,i]
  __shared__ _Float16 kb[C_][DS_ + 8];  // B[u][j] = k[u,j]
  const int b = blockIdx.x / NC_;
  const int c = blockIdx.x % NC_;
  const size_t base = ((size_t)b * T_ + (size_t)c * C_) * DS_;

  for (int idx = threadIdx.x; idx < C_ * DS_; idx += 128) {
    const int u = idx >> 6, i = idx & 63;
    const float dd = sigmoidf_(decay[i]);
    const float sc = __powf(dd, (float)(C_ - 1 - u));
    aT[i][u] = (_Float16)(v[base + (size_t)u * DS_ + i] * sc);
    kb[u][i] = (_Float16)k[base + (size_t)u * DS_ + i];
  }
  __syncthreads();

  const int wave = threadIdx.x >> 5;  // i-tile
  const int lane = threadIdx.x & 31;
  const int nn = lane & 15, h = lane >> 4;
  const v16h a0 = afrag16(&aT[wave * 16][0], C_ + 8, 0);
  const v16h a1 = afrag16(&aT[wave * 16][0], C_ + 8, 32);
  for (int jt = 0; jt < 4; ++jt) {
    v8f acc = {};
    const v16h b0 = bfrag16(&kb[0][0], DS_ + 8, 0, jt * 16);
    const v16h b1 = bfrag16(&kb[0][0], DS_ + 8, 32, jt * 16);
    acc = wmma_f16(a0, b0, acc);
    acc = wmma_f16(a1, b1, acc);
#pragma unroll
    for (int g = 0; g < 8; ++g) {
      const int i = wave * 16 + g + (h << 3);
      const int j = jt * 16 + nn;
      G[((size_t)blockIdx.x * DS_ + i) * DS_ + j] = acc[g];
    }
  }
}

// ================= Kernel 3: tiny sequential inter-chunk scan =================
// S_{c+1} = diag(d^C) S_c + G_c ; stores pre-chunk state S_c for every chunk.
__global__ __launch_bounds__(256) void state_kernel(
    const float* __restrict__ G, const float* __restrict__ decay,
    float* __restrict__ S) {
  const int b = blockIdx.x;
  const int tid = threadIdx.x;
  const int i = tid >> 2;  // 16 consecutive elems share row i
  const float dd = sigmoidf_(decay[i]);
  const float dC = __powf(dd, (float)C_);
  float s[16];
#pragma unroll
  for (int e = 0; e < 16; ++e) s[e] = 0.f;
  const int flat = tid * 16;
  for (int c = 0; c < NC_; ++c) {
    const size_t off = (size_t)(b * NC_ + c) * (DS_ * DS_) + flat;
#pragma unroll
    for (int e = 0; e < 16; ++e) S[off + e] = s[e];
#pragma unroll
    for (int e = 0; e < 16; ++e) s[e] = dC * s[e] + G[off + e];
  }
}

// ================= Kernel 4: per-chunk attention + output proj + residual =====
// o = diag(d^r) * (Mask(Q K^T) @ Vt) + diag(d^{r+1}) o (Q S_c^T)
// out = x + o @ Wo^T + bo
__global__ __launch_bounds__(256) void out_kernel(
    const float* __restrict__ x, const float* __restrict__ q,
    const float* __restrict__ k, const float* __restrict__ v,
    const float* __restrict__ decay, const float* __restrict__ S,
    const float* __restrict__ Wo, const float* __restrict__ bo,
    float* __restrict__ out) {
  __shared__ _Float16 qs[C_][DS_ + 8];   // Q[r][s]
  __shared__ _Float16 ST[DS_][DS_ + 8];  // B for O2: ST[j][i] = S[i][j]
  __shared__ float    Vt[C_][DS_ + 8];   // f32 B: Vt[u][i] = d_i^{-u} v[u,i]
  __shared__ float    Pm[C_][C_ + 4];    // masked scores (f32)
  __shared__ union {                     // stage1 kT / stage2+3 o (time-disjoint)
    _Float16 kT[DS_][C_ + 8];            // B for P: kT[s][u] = k[u,s]
    _Float16 os[C_][DS_ + 8];            // o[r][i] as f16
  } ubuf;
  __shared__ float l2d[DS_];

  const int b = blockIdx.x / NC_;
  const int c = blockIdx.x % NC_;
  const size_t base = ((size_t)b * T_ + (size_t)c * C_) * DS_;
  const size_t sbase = (size_t)(b * NC_ + c) * (DS_ * DS_);
  const size_t xbase = ((size_t)b * T_ + (size_t)c * C_) * D_;
  const int tid = threadIdx.x;

  // prefetch the residual x tile (64x1024 f32 = 2048 cachelines)
#pragma unroll
  for (int pp = 0; pp < 8; ++pp)
    __builtin_prefetch(&x[xbase + (size_t)(tid * 8 + pp) * 32], 0, 1);

  if (tid < DS_) l2d[tid] = __log2f(sigmoidf_(decay[tid]));
  __syncthreads();

  for (int idx = tid; idx < C_ * DS_; idx += 256) {
    const int rr = idx >> 6, cc = idx & 63;
    qs[rr][cc] = (_Float16)q[base + (size_t)rr * DS_ + cc];
    ubuf.kT[cc][rr] = (_Float16)k[base + (size_t)rr * DS_ + cc];
    ST[cc][rr] = (_Float16)S[sbase + (size_t)rr * DS_ + cc];
    Vt[rr][cc] = v[base + (size_t)rr * DS_ + cc] * exp2f(-(float)rr * l2d[cc]);
  }
  __syncthreads();

  const int wave = tid >> 5, lane = tid & 31;
  const int nn = lane & 15, h = lane >> 4;

  // ---- stage 1: P = causal(Q K^T), 16 tiles / 8 waves ----
  for (int tt = wave * 2; tt < wave * 2 + 2; ++tt) {
    const int rt = tt >> 2, ut = tt & 3;
    if (ut > rt) {
#pragma unroll
      for (int g = 0; g < 8; ++g) Pm[rt * 16 + g + (h << 3)][ut * 16 + nn] = 0.f;
      continue;
    }
    const v16h pa0 = afrag16(&qs[rt * 16][0], DS_ + 8, 0);
    const v16h pa1 = afrag16(&qs[rt * 16][0], DS_ + 8, 32);
    const v16h pb0 = bfrag16(&ubuf.kT[0][0], C_ + 8, 0, ut * 16);
    const v16h pb1 = bfrag16(&ubuf.kT[0][0], C_ + 8, 32, ut * 16);
    v8f acc = {};
    acc = wmma_f16(pa0, pb0, acc);
    acc = wmma_f16(pa1, pb1, acc);
#pragma unroll
    for (int g = 0; g < 8; ++g) {
      const int r = rt * 16 + g + (h << 3);
      const int u = ut * 16 + nn;
      Pm[r][u] = (u <= r) ? acc[g] : 0.f;
    }
  }
  __syncthreads();

  // ---- stage 2: o = d^r*(P@Vt) + d^{r+1}*(Q@S^T), f32 + f16 WMMA ----
  for (int tt = wave * 2; tt < wave * 2 + 2; ++tt) {
    const int rt = tt >> 2, it = tt & 3;
    v8f o1 = {};
#pragma unroll
    for (int k4 = 0; k4 < C_; k4 += 4) {
      const v2f fa = afrag32(&Pm[rt * 16][0], C_ + 4, k4);
      const v2f fb = bfrag32(&Vt[0][0], DS_ + 8, k4, it * 16);
      o1 = wmma_f32(fa, fb, o1);
    }
    v8f o2 = {};
    const v16h qa0 = afrag16(&qs[rt * 16][0], DS_ + 8, 0);
    const v16h qa1 = afrag16(&qs[rt * 16][0], DS_ + 8, 32);
    const v16h sb0 = bfrag16(&ST[0][0], DS_ + 8, 0, it * 16);
    const v16h sb1 = bfrag16(&ST[0][0], DS_ + 8, 32, it * 16);
    o2 = wmma_f16(qa0, sb0, o2);
    o2 = wmma_f16(qa1, sb1, o2);
#pragma unroll
    for (int g = 0; g < 8; ++g) {
      const int r = rt * 16 + g + (h << 3);
      const int i = it * 16 + nn;
      const float ld_ = l2d[i];
      const float dr = exp2f((float)r * ld_);
      const float di = exp2f(ld_);
      ubuf.os[r][i] = (_Float16)(dr * (o1[g] + di * o2[g]));
    }
  }
  __syncthreads();

  // ---- stage 3: out = x + o @ Wo^T + bo ; 256 tiles / 8 waves ----
  const int rt = wave & 3;
  const int dt0 = (wave >> 2) * 32;
  const v16h oa0 = afrag16(&ubuf.os[rt * 16][0], DS_ + 8, 0);
  const v16h oa1 = afrag16(&ubuf.os[rt * 16][0], DS_ + 8, 32);
  for (int dt = dt0; dt < dt0 + 32; ++dt) {
    const int n0 = dt * 16;
    v16h wb0, wb1;
#pragma unroll
    for (int e = 0; e < 16; ++e) {  // B[s][dcol] = Wo[dcol][s]
      wb0[e] = (_Float16)Wo[(size_t)(n0 + nn) * DS_ + h * 16 + e];
      wb1[e] = (_Float16)Wo[(size_t)(n0 + nn) * DS_ + 32 + h * 16 + e];
    }
    v8f acc = {};
    acc = wmma_f16(oa0, wb0, acc);
    acc = wmma_f16(oa1, wb1, acc);
    const float bias = bo[n0 + nn];
#pragma unroll
    for (int g = 0; g < 8; ++g) {
      const int r = rt * 16 + g + (h << 3);
      const size_t oidx = xbase + (size_t)r * D_ + n0 + nn;
      out[oidx] = x[oidx] + acc[g] + bias;
    }
  }
}

// =============================== launcher ===============================
extern "C" void kernel_launch(void* const* d_in, const int* in_sizes, int n_in,
                              void* d_out, int out_size, void* d_ws,
                              size_t ws_size, hipStream_t stream) {
  (void)in_sizes; (void)n_in; (void)out_size; (void)ws_size;
  const float* x     = (const float*)d_in[0];
  const float* Wq    = (const float*)d_in[1];
  const float* bq    = (const float*)d_in[2];
  const float* Wk    = (const float*)d_in[3];
  const float* bk    = (const float*)d_in[4];
  const float* Wv    = (const float*)d_in[5];
  const float* bv    = (const float*)d_in[6];
  const float* decay = (const float*)d_in[7];
  const float* Wo    = (const float*)d_in[8];
  const float* bo    = (const float*)d_in[9];
  float* out = (float*)d_out;

  const size_t NQKV = (size_t)B_ * T_ * DS_;          // 2,097,152
  const size_t NG   = (size_t)B_ * NC_ * DS_ * DS_;   // 2,097,152
  float* qb = (float*)d_ws;
  float* kb = qb + NQKV;
  float* vb = kb + NQKV;
  float* Gb = vb + NQKV;
  float* Sb = Gb + NG;

  qkv_kernel<<<(B_ * T_) / 16, 128, 0, stream>>>(x, Wq, bq, Wk, bk, Wv, bv,
                                                 qb, kb, vb);
  chunkG_kernel<<<B_ * NC_, 128, 0, stream>>>(kb, vb, decay, Gb);
  state_kernel<<<B_, 256, 0, stream>>>(Gb, decay, Sb);
  out_kernel<<<B_ * NC_, 256, 0, stream>>>(x, qb, kb, vb, decay, Sb, Wo, bo,
                                           out);
}
